// SelfAttentive_63402307224251
// MI455X (gfx1250) — matrix operand order repaired and checked by reading, not compiled, not run on hardware
//
#include <hip/hip_runtime.h>
#include <hip/hip_bf16.h>
#include <math.h>

// Problem dims
#define B_   64
#define T_   512
#define D_   300
#define DP_  320     // D padded to mult of 32 (WMMA K step)
#define H_   512
#define H2_  1024    // 2H
#define H3_  1536    // 3H
#define DA_  350
#define DAP_ 352     // DA padded
#define R_   30
#define RP_  32      // R padded

typedef __attribute__((ext_vector_type(16))) __bf16 v16bf;
typedef __attribute__((ext_vector_type(8)))  float  v8f;

union FragBF { v16bf v; uint4 u[2]; };

__device__ __forceinline__ unsigned short f2bf(float f) {
  unsigned int u = __float_as_uint(f);
  unsigned int r = (u + 0x7FFFu + ((u >> 16) & 1u)) >> 16;
  return (unsigned short)r;
}
__device__ __forceinline__ float sigm(float x) { return 1.0f / (1.0f + expf(-x)); }

// ---- WMMA fragment loaders (CDNA5 wave32 16-bit layouts) ----
// A 16x32 (MxK): lane<16 -> M=lane, K runs [k0+0..7],[k0+16..23];
//                lane>=16 -> M=lane-16, K runs [k0+8..15],[k0+24..31]
__device__ __forceinline__ v16bf load_fragA(const unsigned short* A, int lda,
                                            int m0, int k0, int lane) {
  FragBF f;
  int half = lane >> 4;
  int m = m0 + (lane & 15);
  const unsigned short* base = A + (size_t)m * lda + k0 + half * 8;
  f.u[0] = *(const uint4*)(base);
  f.u[1] = *(const uint4*)(base + 16);
  return f.v;
}
// B 32x16 (KxN): lane<16 -> N=lane, K runs [k0+0..7],[k0+16..23]; lane>=16 -> N=lane-16, K runs [k0+8..15],[k0+24..31]
// Bw stored row-major [N, ldb] over K (i.e. weight rows contiguous in K).
__device__ __forceinline__ v16bf load_fragB(const unsigned short* Bw, int ldb,
                                            int n0, int k0, int lane) {
  FragBF f;
  int half = lane >> 4;
  int n = n0 + (lane & 15);
  const unsigned short* base = Bw + (size_t)n * ldb + k0 + half * 8;
  f.u[0] = *(const uint4*)(base);
  f.u[1] = *(const uint4*)(base + 16);
  return f.v;
}

__device__ __forceinline__ v8f vzero8() {
  v8f z = {0.f, 0.f, 0.f, 0.f, 0.f, 0.f, 0.f, 0.f};
  return z;
}

// ---- Generic tiled WMMA GEMM: C[m,n] = sum_k A[m,k] * W[n,k] (+ bias[n]) ----
// MODE 0: f32 store (+optional bias). MODE 1: tanh -> bf16 store.
// Block: 128 threads (4 waves). Block covers 16 rows x 256 cols; wave covers 64 cols.
template <int MODE>
__global__ void gemm_wmma_kernel(const unsigned short* __restrict__ A, int lda,
                                 const unsigned short* __restrict__ Bw, int ldb,
                                 void* __restrict__ Cout, int ldc,
                                 int Nreal, int Npad, int K,
                                 const float* __restrict__ bias) {
  int lane = threadIdx.x & 31;
  int wave = threadIdx.x >> 5;
  int m0 = blockIdx.x * 16;
  int nbase = blockIdx.y * 256 + wave * 64;

  v8f acc[4];
#pragma unroll
  for (int tt = 0; tt < 4; ++tt) acc[tt] = vzero8();

  for (int k0 = 0; k0 < K; k0 += 32) {
    v16bf fa = load_fragA(A, lda, m0, k0, lane);
#pragma unroll
    for (int tt = 0; tt < 4; ++tt) {
      int n0 = nbase + tt * 16;
      if (n0 >= Npad) continue;  // wave-uniform
      v16bf fb = load_fragB(Bw, ldb, n0, k0, lane);
      acc[tt] = __builtin_amdgcn_wmma_f32_16x16x32_bf16(
          false, fa, false, fb, (short)0, acc[tt], false, false);
    }
  }

  int half = lane >> 4;
#pragma unroll
  for (int tt = 0; tt < 4; ++tt) {
    int n0 = nbase + tt * 16;
    if (n0 >= Npad) continue;
    int c = n0 + (lane & 15);
    if (c >= Nreal) continue;
    int rbase = m0 + half * 8;
    if (MODE == 0) {
      float bv = bias ? bias[c] : 0.0f;
      float* C = (float*)Cout;
#pragma unroll
      for (int v2 = 0; v2 < 8; ++v2)
        C[(size_t)(rbase + v2) * ldc + c] = acc[tt][v2] + bv;
    } else {
      unsigned short* C = (unsigned short*)Cout;
#pragma unroll
      for (int v2 = 0; v2 < 8; ++v2)
        C[(size_t)(rbase + v2) * ldc + c] = f2bf(tanhf(acc[tt][v2]));
    }
  }
}

// ---- Fused GRU step: hg = h @ Whh^T (WMMA) + gates + state update ----
// Grid: 32 blocks (16 cols of H each), 128 threads (4 waves, 16 batch rows each).
__global__ void gru_step_kernel(const unsigned short* __restrict__ hbf_in,
                                const float* __restrict__ hf_in,
                                float* __restrict__ hf_out,
                                unsigned short* __restrict__ hbf_out,
                                const unsigned short* __restrict__ Whh_bf, // [3H,512] bf16
                                const float* __restrict__ bhh,
                                const float* __restrict__ xg,              // [B*T,3H] f32
                                const int* __restrict__ lengths,
                                float* __restrict__ HoutF,                 // [B*T,2H]
                                unsigned short* __restrict__ HoutB,
                                int t, int backward) {
  int lane = threadIdx.x & 31;
  int wave = threadIdx.x >> 5;
  int j = blockIdx.x * 16;   // column tile within H
  int m0 = wave * 16;        // batch-row tile

  v8f accR = vzero8(), accZ = vzero8(), accN = vzero8();
  for (int k0 = 0; k0 < H_; k0 += 32) {
    v16bf fa = load_fragA(hbf_in, H_, m0, k0, lane);
    v16bf fr = load_fragB(Whh_bf, H_, j, k0, lane);
    accR = __builtin_amdgcn_wmma_f32_16x16x32_bf16(false, fa, false, fr, (short)0, accR, false, false);
    v16bf fz = load_fragB(Whh_bf, H_, H_ + j, k0, lane);
    accZ = __builtin_amdgcn_wmma_f32_16x16x32_bf16(false, fa, false, fz, (short)0, accZ, false, false);
    v16bf fn = load_fragB(Whh_bf, H_, 2 * H_ + j, k0, lane);
    accN = __builtin_amdgcn_wmma_f32_16x16x32_bf16(false, fa, false, fn, (short)0, accN, false, false);
  }

  int half = lane >> 4;
  int c = j + (lane & 15);
  float br = bhh[c], bz = bhh[H_ + c], bn = bhh[2 * H_ + c];
#pragma unroll
  for (int v2 = 0; v2 < 8; ++v2) {
    int m = m0 + half * 8 + v2;  // batch index
    const float* xrow = xg + ((size_t)m * T_ + t) * H3_;
    float r = sigm(xrow[c] + accR[v2] + br);
    float z = sigm(xrow[H_ + c] + accZ[v2] + bz);
    float n = tanhf(xrow[2 * H_ + c] + r * (accN[v2] + bn));
    float hold = hf_in[m * H_ + c];
    float hnew = (1.0f - z) * n + z * hold;
    int len = lengths[m];
    float hcur = (t < len) ? hnew : hold;   // freeze past length
    hf_out[m * H_ + c] = hcur;
    hbf_out[m * H_ + c] = f2bf(hcur);
    if (t < len) {
      int tp = backward ? (len - 1 - t) : t; // un-reverse backward outputs
      size_t oi = ((size_t)m * T_ + tp) * H2_ + (backward ? H_ : 0) + c;
      HoutF[oi] = hcur;
      HoutB[oi] = f2bf(hcur);
    }
  }
}

// ---- Embedding gather (forward + per-length-reversed) -> bf16, K-padded ----
__global__ void embed_kernel(const int* __restrict__ input, const int* __restrict__ lengths,
                             const float* __restrict__ embm,
                             unsigned short* __restrict__ embF,
                             unsigned short* __restrict__ embB) {
  size_t idx = (size_t)blockIdx.x * blockDim.x + threadIdx.x;
  size_t total = (size_t)B_ * T_ * DP_;
  if (idx >= total) return;
  int d = (int)(idx % DP_);
  size_t m = idx / DP_;
  int b = (int)(m / T_), t = (int)(m % T_);
  float vf = 0.f, vb = 0.f;
  if (d < D_) {
    int tok = input[b * T_ + t];
    vf = embm[(size_t)tok * D_ + d];
    int len = lengths[b];
    int tr = len - 1 - t;
    if (tr < 0) tr = 0;
    if (tr > T_ - 1) tr = T_ - 1;
    int tokr = input[b * T_ + tr];
    vb = embm[(size_t)tokr * D_ + d];
  }
  embF[idx] = f2bf(vf);
  embB[idx] = f2bf(vb);
}

// ---- f32 -> bf16 with zero padding ----
__global__ void cvt_pad_kernel(const float* __restrict__ src, unsigned short* __restrict__ dst,
                               int rows, int cols, int rowsP, int colsP) {
  size_t idx = (size_t)blockIdx.x * blockDim.x + threadIdx.x;
  size_t total = (size_t)rowsP * colsP;
  if (idx >= total) return;
  int r = (int)(idx / colsP), c = (int)(idx % colsP);
  float v = (r < rows && c < cols) ? src[(size_t)r * cols + c] : 0.0f;
  dst[idx] = f2bf(v);
}

__global__ void hinit_kernel(const float* __restrict__ hidden,
                             float* hfF, float* hfB,
                             unsigned short* hbF, unsigned short* hbB) {
  int idx = blockIdx.x * blockDim.x + threadIdx.x;
  if (idx >= B_ * H_) return;
  float vf = hidden[idx];
  float vb = hidden[B_ * H_ + idx];
  hfF[idx] = vf; hfB[idx] = vb;
  hbF[idx] = f2bf(vf); hbB[idx] = f2bf(vb);
}

__global__ void ht_copy_kernel(const float* __restrict__ hF, const float* __restrict__ hB,
                               float* __restrict__ out) {
  int idx = blockIdx.x * blockDim.x + threadIdx.x;
  if (idx >= B_ * H_) return;
  out[idx] = hF[idx];
  out[B_ * H_ + idx] = hB[idx];
}

// ---- softmax over T per (b,r); masked positions -> 0; writes A [B,T,R] ----
__global__ void softmax_kernel(const float* __restrict__ s2, const int* __restrict__ lengths,
                               float* __restrict__ A) {
  int br = blockIdx.x;
  int b = br / R_, r = br % R_;
  int len = lengths[b];
  int tid = threadIdx.x;
  __shared__ float red[256];

  float mx = -INFINITY;
  for (int t = tid; t < T_; t += 256)
    if (t < len) mx = fmaxf(mx, s2[((size_t)b * T_ + t) * RP_ + r]);
  red[tid] = mx; __syncthreads();
  for (int s = 128; s > 0; s >>= 1) { if (tid < s) red[tid] = fmaxf(red[tid], red[tid + s]); __syncthreads(); }
  mx = red[0]; __syncthreads();

  float sum = 0.f;
  for (int t = tid; t < T_; t += 256)
    if (t < len) sum += expf(s2[((size_t)b * T_ + t) * RP_ + r] - mx);
  red[tid] = sum; __syncthreads();
  for (int s = 128; s > 0; s >>= 1) { if (tid < s) red[tid] += red[tid + s]; __syncthreads(); }
  sum = red[0];

  float inv = 1.0f / sum;
  for (int t = tid; t < T_; t += 256) {
    float a = (t < len) ? expf(s2[((size_t)b * T_ + t) * RP_ + r] - mx) * inv : 0.0f;
    A[((size_t)b * T_ + t) * R_ + r] = a;
  }
}

// ---- M[b,r,d] = sum_t A[b,t,r] * Hout[b,t,d]; writes BM directly ----
__global__ void attn_m_kernel(const float* __restrict__ A, const float* __restrict__ Hout,
                              float* __restrict__ BM) {
  int b = blockIdx.y;
  int d = blockIdx.x * 256 + threadIdx.x;  // < 1024
  __shared__ float As[R_];
  float acc[R_];
#pragma unroll
  for (int r = 0; r < R_; ++r) acc[r] = 0.f;
  for (int t = 0; t < T_; ++t) {
    __syncthreads();
    if (threadIdx.x < R_) As[threadIdx.x] = A[((size_t)b * T_ + t) * R_ + threadIdx.x];
    __syncthreads();
    float h = Hout[((size_t)b * T_ + t) * H2_ + d];
#pragma unroll
    for (int r = 0; r < R_; ++r) acc[r] += As[r] * h;
  }
#pragma unroll
  for (int r = 0; r < R_; ++r)
    BM[(size_t)b * (R_ * H2_) + (size_t)r * H2_ + d] = acc[r];
}

// ---- penal partials: per-b sum over (r,s) of (AAT - I)^2 ----
__global__ void penal_partial_kernel(const float* __restrict__ A, float* __restrict__ part) {
  int b = blockIdx.x;
  int tid = threadIdx.x;
  __shared__ float As[R_];
  __shared__ float red[256];
  float acc[4] = {0.f, 0.f, 0.f, 0.f};
  for (int t = 0; t < T_; ++t) {
    __syncthreads();
    if (tid < R_) As[tid] = A[((size_t)b * T_ + t) * R_ + tid];
    __syncthreads();
#pragma unroll
    for (int q = 0; q < 4; ++q) {
      int p = tid + q * 256;
      if (p < R_ * R_) {
        int r = p / R_, s = p % R_;
        acc[q] += As[r] * As[s];
      }
    }
  }
  float loc = 0.f;
#pragma unroll
  for (int q = 0; q < 4; ++q) {
    int p = tid + q * 256;
    if (p < R_ * R_) {
      int r = p / R_, s = p % R_;
      float dv = acc[q] - (r == s ? 1.0f : 0.0f);
      loc += dv * dv;
    }
  }
  red[tid] = loc; __syncthreads();
  for (int s = 128; s > 0; s >>= 1) { if (tid < s) red[tid] += red[tid + s]; __syncthreads(); }
  if (tid == 0) part[b] = red[0];
}

__global__ void penal_final_kernel(const float* __restrict__ part, float* __restrict__ out) {
  if (threadIdx.x == 0 && blockIdx.x == 0) {
    float s = 0.f;
    for (int b = 0; b < B_; ++b) s += part[b];  // deterministic order
    *out = s / (float)B_;
  }
}

extern "C" void kernel_launch(void* const* d_in, const int* in_sizes, int n_in,
                              void* d_out, int out_size, void* d_ws, size_t ws_size,
                              hipStream_t stream) {
  (void)in_sizes; (void)n_in; (void)out_size; (void)ws_size;
  const int*   input   = (const int*)  d_in[0];
  const float* hidden  = (const float*)d_in[1];
  const int*   lengths = (const int*)  d_in[2];
  const float* embm    = (const float*)d_in[3];
  const float* Wih_f   = (const float*)d_in[4];
  const float* Whh_f   = (const float*)d_in[5];
  const float* bih_f   = (const float*)d_in[6];
  const float* bhh_f   = (const float*)d_in[7];
  const float* Wih_b   = (const float*)d_in[8];
  const float* Whh_b   = (const float*)d_in[9];
  const float* bih_b   = (const float*)d_in[10];
  const float* bhh_b   = (const float*)d_in[11];
  const float* W1      = (const float*)d_in[12];
  const float* W2      = (const float*)d_in[13];
  float* out = (float*)d_out;

  const size_t MT = (size_t)B_ * T_;  // 32768 rows

  // ---- workspace carving (256B aligned) ----
  char* p = (char*)d_ws;
  auto alloc = [&](size_t bytes) -> char* {
    char* r = p;
    p += (bytes + 255) & ~(size_t)255;
    return r;
  };
  unsigned short* embF   = (unsigned short*)alloc(MT * DP_ * 2);
  unsigned short* embB   = (unsigned short*)alloc(MT * DP_ * 2);
  unsigned short* WihFb  = (unsigned short*)alloc((size_t)H3_ * DP_ * 2);
  unsigned short* WihBb  = (unsigned short*)alloc((size_t)H3_ * DP_ * 2);
  unsigned short* WhhFb  = (unsigned short*)alloc((size_t)H3_ * H_ * 2);
  unsigned short* WhhBb  = (unsigned short*)alloc((size_t)H3_ * H_ * 2);
  unsigned short* W1b    = (unsigned short*)alloc((size_t)DAP_ * H2_ * 2);
  unsigned short* W2b    = (unsigned short*)alloc((size_t)RP_ * DAP_ * 2);
  float*          xgF    = (float*)alloc(MT * H3_ * 4);
  float*          xgB    = (float*)alloc(MT * H3_ * 4);
  float*          HoutF  = (float*)alloc(MT * H2_ * 4);
  unsigned short* HoutBf = (unsigned short*)alloc(MT * H2_ * 2);
  unsigned short* ts1    = (unsigned short*)alloc(MT * DAP_ * 2);
  float*          s2f    = (float*)alloc(MT * RP_ * 4);
  float*          hfF[2], *hfB[2];
  unsigned short* hbF[2], *hbB[2];
  for (int i = 0; i < 2; ++i) {
    hfF[i] = (float*)alloc((size_t)B_ * H_ * 4);
    hfB[i] = (float*)alloc((size_t)B_ * H_ * 4);
    hbF[i] = (unsigned short*)alloc((size_t)B_ * H_ * 2);
    hbB[i] = (unsigned short*)alloc((size_t)B_ * H_ * 2);
  }
  float* penPart = (float*)alloc((size_t)B_ * 4);

  // ---- output layout ----
  const size_t BM_SZ = (size_t)B_ * R_ * H2_;   // 1,966,080
  const size_t HT_SZ = (size_t)2 * B_ * H_;     // 65,536
  float* outBM  = out;
  float* outHT  = out + BM_SZ;
  float* outPen = out + BM_SZ + HT_SZ;
  float* outA   = out + BM_SZ + HT_SZ + 1;

  // ---- zero-init masked intermediates ----
  hipMemsetAsync(HoutF,  0, MT * H2_ * 4, stream);
  hipMemsetAsync(HoutBf, 0, MT * H2_ * 2, stream);
  hipMemsetAsync(ts1,    0, MT * DAP_ * 2, stream);

  // ---- weight conversions (bf16, zero-padded) ----
  {
    auto launch_cvt = [&](const float* src, unsigned short* dst, int r, int c, int rp, int cp) {
      size_t total = (size_t)rp * cp;
      cvt_pad_kernel<<<(unsigned)((total + 255) / 256), 256, 0, stream>>>(src, dst, r, c, rp, cp);
    };
    launch_cvt(Wih_f, WihFb, H3_, D_,  H3_, DP_);
    launch_cvt(Wih_b, WihBb, H3_, D_,  H3_, DP_);
    launch_cvt(Whh_f, WhhFb, H3_, H_,  H3_, H_);
    launch_cvt(Whh_b, WhhBb, H3_, H_,  H3_, H_);
    launch_cvt(W1,    W1b,   DA_, H2_, DAP_, H2_);
    launch_cvt(W2,    W2b,   R_,  DA_, RP_,  DAP_);
  }

  // ---- embedding gather (fwd + reversed) ----
  {
    size_t total = MT * DP_;
    embed_kernel<<<(unsigned)((total + 255) / 256), 256, 0, stream>>>(input, lengths, embm, embF, embB);
  }

  // ---- input-gate GEMMs: xg = emb @ Wih^T + bih  (WMMA) ----
  {
    dim3 grid((unsigned)(MT / 16), H3_ / 256);
    gemm_wmma_kernel<0><<<grid, 128, 0, stream>>>(embF, DP_, WihFb, DP_, xgF, H3_, H3_, H3_, DP_, bih_f);
    gemm_wmma_kernel<0><<<grid, 128, 0, stream>>>(embB, DP_, WihBb, DP_, xgB, H3_, H3_, H3_, DP_, bih_b);
  }

  // ---- init hidden states (double-buffered) ----
  hinit_kernel<<<(B_ * H_ + 255) / 256, 256, 0, stream>>>(hidden, hfF[0], hfB[0], hbF[0], hbB[0]);

  // ---- recurrence: fused WMMA GEMM + gates, one launch per (t, dir) ----
  for (int t = 0; t < T_; ++t) {
    int cur = t & 1, nxt = cur ^ 1;
    gru_step_kernel<<<H_ / 16, 128, 0, stream>>>(hbF[cur], hfF[cur], hfF[nxt], hbF[nxt],
                                                 WhhFb, bhh_f, xgF, lengths,
                                                 HoutF, HoutBf, t, 0);
    gru_step_kernel<<<H_ / 16, 128, 0, stream>>>(hbB[cur], hfB[cur], hfB[nxt], hbB[nxt],
                                                 WhhBb, bhh_b, xgB, lengths,
                                                 HoutF, HoutBf, t, 1);
  }
  // T_=512 is even -> final state in buffer 0
  ht_copy_kernel<<<(B_ * H_ + 255) / 256, 256, 0, stream>>>(hfF[0], hfB[0], outHT);

  // ---- attention: s1 = tanh(Hout @ W1^T) (WMMA, fused tanh->bf16) ----
  {
    dim3 grid((unsigned)(MT / 16), (DAP_ + 255) / 256);
    gemm_wmma_kernel<1><<<grid, 128, 0, stream>>>(HoutBf, H2_, W1b, H2_, ts1, DAP_,
                                                  DA_, DAP_, H2_, nullptr);
  }
  // ---- s2 = tanh(s1) @ W2^T (WMMA) ----
  {
    dim3 grid((unsigned)(MT / 16), 1);
    gemm_wmma_kernel<0><<<grid, 128, 0, stream>>>(ts1, DAP_, W2b, DAP_, s2f, RP_,
                                                  R_, RP_, DAP_, nullptr);
  }
  // ---- masked softmax over T -> A ----
  softmax_kernel<<<B_ * R_, 256, 0, stream>>>(s2f, lengths, outA);

  // ---- M = A^T Hout -> BM ----
  {
    dim3 grid(H2_ / 256, B_);
    attn_m_kernel<<<grid, 256, 0, stream>>>(outA, HoutF, outBM);
  }

  // ---- penalization ----
  penal_partial_kernel<<<B_, 256, 0, stream>>>(outA, penPart);
  penal_final_kernel<<<1, 1, 0, stream>>>(penPart, outPen);
}
